// PolygonGraphTransformerEncoder_9740985827992
// MI455X (gfx1250) — matrix operand order repaired and checked by reading, not compiled, not run on hardware
//
#include <hip/hip_runtime.h>

// ---------------------------------------------------------------------------
// PolygonGraphTransformerEncoder for gfx1250 (MI455X, wave32, WMMA, async-LDS)
// N=50000 nodes, E=1e6 edges, H=64, HEADS=4, D=16, L=3, B=512, EMB=128
// ---------------------------------------------------------------------------

typedef __attribute__((ext_vector_type(16))) _Float16 v16h;
typedef __attribute__((ext_vector_type(8)))  float    v8f;
typedef int v4i_vs __attribute__((vector_size(16)));

#define HDIM  64
#define NHEAD 4
#define DHEAD 16
#define EMBD  128

// -------------------- CDNA5 async global->LDS staging --------------------

#if __has_builtin(__builtin_amdgcn_global_load_async_to_lds_b128)
#define HAVE_ASYNC_LDS 1
#endif

__device__ __forceinline__ void stage16(const float* gsrc, float* lds_dst) {
#ifdef HAVE_ASYNC_LDS
  __builtin_amdgcn_global_load_async_to_lds_b128(
      (__attribute__((address_space(1))) v4i_vs*)(uintptr_t)gsrc,
      (__attribute__((address_space(3))) v4i_vs*)(uintptr_t)lds_dst,
      0, 0);
#else
  *(float4*)lds_dst = *(const float4*)gsrc;
#endif
}

__device__ __forceinline__ void stage_wait() {
#ifdef HAVE_ASYNC_LDS
#if __has_builtin(__builtin_amdgcn_s_wait_asynccnt)
  __builtin_amdgcn_s_wait_asynccnt(0);
#endif
#endif
}

// -------------------- helpers --------------------

__device__ __forceinline__ void atomic_max_f32(float* addr, float val) {
  if (val >= 0.f) atomicMax((int*)addr, __float_as_int(val));
  else            atomicMin((unsigned int*)addr, __float_as_uint(val));
}

__device__ __forceinline__ float gelu_exact(float y) {
  return 0.5f * y * (1.0f + erff(y * 0.70710678118654752f));
}

// -------------------- utility kernels --------------------

__global__ void fill_kernel(float* __restrict__ p, float v, int n) {
  int i = blockIdx.x * blockDim.x + threadIdx.x;
  if (i < n) p[i] = v;
}

__global__ void fix_neginf_kernel(float* __restrict__ p, int n) {
  int i = blockIdx.x * blockDim.x + threadIdx.x;
  if (i < n) { float v = p[i]; if (v == -__builtin_inff()) p[i] = 0.f; }
}

// -------------------- weight repack: f32 row-major -> f16 B-fragment layout ----
// Consumer: fragment (tile t, K-chunk c), lane L, element j holds
//   W[(c*32 + (j<8 ? (L>>4)*8 + j : 16 + (L>>4)*8 + (j-8))) * Ncols + t*16 + (L&15)]
// Packed offset: ((t*nc + c)*32 + L)*16 + j  -> each lane's fragment is 32B.

__global__ void pack_b_kernel(const float* __restrict__ W, _Float16* __restrict__ P,
                              int K, int Ncols) {
  const int nc = K / 32;
  const int total = (Ncols / 16) * nc * 32;
  int i = blockIdx.x * blockDim.x + threadIdx.x;
  if (i >= total) return;
  const int lane  = i & 31;
  const int chunk = (i >> 5) % nc;
  const int tile  = (i >> 5) / nc;
  const int group = lane >> 4;
  const int col   = tile * 16 + (lane & 15);
  _Float16* dst = P + (size_t)i * 16;
#pragma unroll
  for (int j = 0; j < 16; j++)
    dst[j] = (_Float16)W[(chunk * 32 + (j < 8 ? group * 8 + j : 16 + group * 8 + (j - 8))) * Ncols + col];
}

// -------------------- input projection: x[N,8] @ W_in[8,64] + b --------------------

__global__ void input_proj_kernel(const float* __restrict__ x8,
                                  const float* __restrict__ W,
                                  const float* __restrict__ b,
                                  float* __restrict__ x64, int n) {
  int i = blockIdx.x * blockDim.x + threadIdx.x;
  if (i >= n * HDIM) return;
  int node = i >> 6, f = i & 63;
  const float* xp = x8 + node * 8;
  float acc = b[f];
#pragma unroll
  for (int j = 0; j < 8; j++) acc += xp[j] * W[j * HDIM + f];
  x64[i] = acc;
}

// -------------------- shared LN-tile staging (device inline) --------------------
// Loads 16x64 f32 tile via async-LDS, computes LN into f16 tile hs.

__device__ __forceinline__ void ln_tile(const float* __restrict__ x, int node0, int n,
                                        const float* __restrict__ g,
                                        const float* __restrict__ bln,
                                        float (*xs)[64], _Float16 (*hs)[72],
                                        float* mus, float* rstds, int tid) {
  const bool full = (node0 + 16 <= n);
  if (full) {
    const float* src = x + (size_t)node0 * HDIM;
#pragma unroll
    for (int it = 0; it < 2; it++) {
      int e = tid + it * 128;          // 256 x 16B transfers
      int row = e >> 4, c4 = (e & 15) * 4;
      stage16(src + row * HDIM + c4, &xs[row][c4]);
    }
    stage_wait();
  } else {
#pragma unroll
    for (int it = 0; it < 2; it++) {
      int e = tid + it * 128;
      int row = e >> 4, c4 = (e & 15) * 4;
      if (node0 + row < n) {
        const float* s = x + (size_t)(node0 + row) * HDIM + c4;
        xs[row][c4] = s[0]; xs[row][c4 + 1] = s[1];
        xs[row][c4 + 2] = s[2]; xs[row][c4 + 3] = s[3];
      } else {
        xs[row][c4] = 0.f; xs[row][c4 + 1] = 0.f;
        xs[row][c4 + 2] = 0.f; xs[row][c4 + 3] = 0.f;
      }
    }
  }
  __syncthreads();
  if (tid < 16) {
    float mu = 0.f;
    for (int j = 0; j < HDIM; j++) mu += xs[tid][j];
    mu *= (1.f / HDIM);
    float var = 0.f;
    for (int j = 0; j < HDIM; j++) { float d = xs[tid][j] - mu; var += d * d; }
    var *= (1.f / HDIM);
    mus[tid] = mu;
    rstds[tid] = rsqrtf(var + 1e-5f);
  }
  __syncthreads();
#pragma unroll
  for (int it = 0; it < 8; it++) {     // 1024 elements / 128 threads
    int e = tid + it * 128;
    int row = e >> 6, col = e & 63;
    hs[row][col] = (_Float16)((xs[row][col] - mus[row]) * rstds[row] * g[col] + bln[col]);
  }
  __syncthreads();
}

// -------------------- fused LN1 + {Q,K,V,Skip} GEMMs (WMMA) --------------------
// 128 threads = 4 waves, 16 nodes per block; wave w -> one output matrix.

__global__ __launch_bounds__(128)
void ln_qkv_skip_kernel(const float* __restrict__ x,
                        const float* __restrict__ g, const float* __restrict__ bln,
                        const _Float16* __restrict__ Pq, const float* __restrict__ bq,
                        const _Float16* __restrict__ Pk, const float* __restrict__ bk,
                        const _Float16* __restrict__ Pv, const float* __restrict__ bv,
                        const _Float16* __restrict__ Ps, const float* __restrict__ bs,
                        float* __restrict__ qo, float* __restrict__ ko,
                        float* __restrict__ vo, float* __restrict__ xro, int n) {
  __shared__ float    xs[16][64];
  __shared__ _Float16 hs[16][72];
  __shared__ float    mus[16], rstds[16];
  const int node0 = blockIdx.x * 16;
  const int tid = threadIdx.x;

  ln_tile(x, node0, n, g, bln, xs, hs, mus, rstds, tid);

  const int wave = tid >> 5;
  const int lane = tid & 31;
  const int laneRow = lane & 15;
  const int laneHi  = lane >> 4;

  const v16h* Wp; const float* bb; float* op;
  if      (wave == 0) { Wp = (const v16h*)Pq; bb = bq; op = qo;  }
  else if (wave == 1) { Wp = (const v16h*)Pk; bb = bk; op = ko;  }
  else if (wave == 2) { Wp = (const v16h*)Pv; bb = bv; op = vo;  }
  else                { Wp = (const v16h*)Ps; bb = bs; op = xro; }

  // A fragments (16x32 f16 CDNA5 layout), both K-chunks.
  v16h a0, a1;
#pragma unroll
  for (int j = 0; j < 8; j++) {
    a0[j]     = hs[laneRow][laneHi * 8 + j];
    a0[8 + j] = hs[laneRow][16 + laneHi * 8 + j];
    a1[j]     = hs[laneRow][32 + laneHi * 8 + j];
    a1[8 + j] = hs[laneRow][48 + laneHi * 8 + j];
  }

  const bool full = (node0 + 16 <= n);
#pragma unroll
  for (int t = 0; t < 4; t++) {
    const int col = t * 16 + laneRow;
    v16h b0 = Wp[(t * 2 + 0) * 32 + lane];      // 2x global_load_b128
    v16h b1 = Wp[(t * 2 + 1) * 32 + lane];
    v8f c = {};
    c = __builtin_amdgcn_wmma_f32_16x16x32_f16(false, a0, false, b0, (short)0, c, false, false);
    c = __builtin_amdgcn_wmma_f32_16x16x32_f16(false, a1, false, b1, (short)0, c, false, false);
    const float bias = bb[col];
    float* obase = op + (size_t)(node0 + 8 * laneHi) * HDIM + col;
    if (full) {
#pragma unroll
      for (int r = 0; r < 8; r++) obase[(size_t)r * HDIM] = c[r] + bias;
    } else {
#pragma unroll
      for (int r = 0; r < 8; r++)
        if (node0 + 8 * laneHi + r < n) obase[(size_t)r * HDIM] = c[r] + bias;
    }
  }
}

// -------------------- edge pass 1: scores + segment max --------------------

__global__ void edge_score_kernel(const int* __restrict__ ei,
                                  const float* __restrict__ ea,
                                  const float* __restrict__ We,   // [4,64]
                                  const float* __restrict__ q,
                                  const float* __restrict__ k,
                                  float* __restrict__ s,
                                  float* __restrict__ m, int nE) {
  int idx = blockIdx.x * blockDim.x + threadIdx.x;
  if (idx >= nE * NHEAD) return;
  int e = idx >> 2, head = idx & 3;
  int src = ei[e], dst = ei[nE + e];
  float a0 = ea[e * 4 + 0], a1 = ea[e * 4 + 1], a2 = ea[e * 4 + 2], a3 = ea[e * 4 + 3];
  const float* qp = q + (size_t)dst * HDIM + head * DHEAD;
  const float* kp = k + (size_t)src * HDIM + head * DHEAD;
  float acc = 0.f;
#pragma unroll
  for (int d = 0; d < DHEAD; d++) {
    int f = head * DHEAD + d;
    float ev = a0 * We[f] + a1 * We[64 + f] + a2 * We[128 + f] + a3 * We[192 + f];
    acc += qp[d] * (kp[d] + ev);
  }
  acc *= 0.25f;  // 1/sqrt(16)
  s[idx] = acc;
  atomic_max_f32(&m[dst * NHEAD + head], acc);
}

// -------------------- edge pass 2: exp + denom + numerator --------------------

__global__ void edge_accum_kernel(const int* __restrict__ ei,
                                  const float* __restrict__ ea,
                                  const float* __restrict__ We,
                                  const float* __restrict__ v,
                                  const float* __restrict__ s,
                                  const float* __restrict__ m,
                                  float* __restrict__ denom,
                                  float* __restrict__ num, int nE) {
  int idx = blockIdx.x * blockDim.x + threadIdx.x;
  if (idx >= nE * NHEAD) return;
  int e = idx >> 2, head = idx & 3;
  int src = ei[e], dst = ei[nE + e];
  float p = expf(s[idx] - m[dst * NHEAD + head]);
  atomicAdd(&denom[dst * NHEAD + head], p);
  float a0 = ea[e * 4 + 0], a1 = ea[e * 4 + 1], a2 = ea[e * 4 + 2], a3 = ea[e * 4 + 3];
  const float* vp = v + (size_t)src * HDIM + head * DHEAD;
  float* np = num + (size_t)dst * HDIM + head * DHEAD;
#pragma unroll
  for (int d = 0; d < DHEAD; d++) {
    int f = head * DHEAD + d;
    float ev = a0 * We[f] + a1 * We[64 + f] + a2 * We[128 + f] + a3 * We[192 + f];
    atomicAdd(&np[d], p * (vp[d] + ev));
  }
}

// -------------------- gate + residual --------------------

__global__ void node_gate_kernel(float* __restrict__ num,
                                 const float* __restrict__ denom,
                                 const float* __restrict__ xr,
                                 const float* __restrict__ Wb,   // [192]
                                 float* __restrict__ x, int n) {
  int node = blockIdx.x * blockDim.x + threadIdx.x;
  if (node >= n) return;
  float* np = num + (size_t)node * HDIM;
  const float* rp = xr + (size_t)node * HDIM;
  float bacc = 0.f;
  for (int f = 0; f < HDIM; f++) {
    int head = f >> 4;
    float o = np[f] / (denom[node * NHEAD + head] + 1e-16f);
    float r = rp[f];
    bacc += o * Wb[f] + r * Wb[64 + f] + (o - r) * Wb[128 + f];
    np[f] = o;
  }
  float beta = 1.f / (1.f + expf(-bacc));
  float* xp = x + (size_t)node * HDIM;
  for (int f = 0; f < HDIM; f++)
    xp[f] += beta * rp[f] + (1.f - beta) * np[f];
}

// -------------------- fused LN2 + FFN (WMMA): x += W2(gelu(W1(ln(x)))) --------

__global__ __launch_bounds__(128)
void ln_ffn_kernel(float* __restrict__ x,
                   const float* __restrict__ g, const float* __restrict__ bln,
                   const _Float16* __restrict__ P1, const float* __restrict__ b1,
                   const _Float16* __restrict__ P2, const float* __restrict__ b2, int n) {
  __shared__ float    xs[16][64];
  __shared__ _Float16 hs[16][72];
  __shared__ _Float16 h2[16][136];
  __shared__ float    mus[16], rstds[16];
  const int node0 = blockIdx.x * 16;
  const int tid = threadIdx.x;

  ln_tile(x, node0, n, g, bln, xs, hs, mus, rstds, tid);

  const int wave = tid >> 5;
  const int lane = tid & 31;
  const int laneRow = lane & 15;
  const int laneHi  = lane >> 4;
  const v16h* W1p = (const v16h*)P1;   // nt=8, nc=2
  const v16h* W2p = (const v16h*)P2;   // nt=4, nc=4

  // ---- GEMM1: [16,64] @ [64,128], gelu -> h2 ----
  v16h a0, a1;
#pragma unroll
  for (int j = 0; j < 8; j++) {
    a0[j]     = hs[laneRow][laneHi * 8 + j];
    a0[8 + j] = hs[laneRow][16 + laneHi * 8 + j];
    a1[j]     = hs[laneRow][32 + laneHi * 8 + j];
    a1[8 + j] = hs[laneRow][48 + laneHi * 8 + j];
  }
#pragma unroll
  for (int i = 0; i < 2; i++) {
    const int t = wave * 2 + i;
    const int col = t * 16 + laneRow;
    v16h b0 = W1p[(t * 2 + 0) * 32 + lane];
    v16h b1v = W1p[(t * 2 + 1) * 32 + lane];
    v8f c = {};
    c = __builtin_amdgcn_wmma_f32_16x16x32_f16(false, a0, false, b0, (short)0, c, false, false);
    c = __builtin_amdgcn_wmma_f32_16x16x32_f16(false, a1, false, b1v, (short)0, c, false, false);
    const float bias = b1[col];
#pragma unroll
    for (int r = 0; r < 8; r++)
      h2[r + 8 * laneHi][col] = (_Float16)gelu_exact(c[r] + bias);
  }
  __syncthreads();

  // ---- GEMM2: [16,128] @ [128,64], residual add from staged xs ----
  const int col = wave * 16 + laneRow;
  v8f c = {};
#pragma unroll
  for (int kc = 0; kc < 4; kc++) {
    v16h a;
#pragma unroll
    for (int j = 0; j < 8; j++) {
      a[j]     = h2[laneRow][kc * 32 + laneHi * 8 + j];
      a[8 + j] = h2[laneRow][kc * 32 + 16 + laneHi * 8 + j];
    }
    v16h b = W2p[(wave * 4 + kc) * 32 + lane];
    c = __builtin_amdgcn_wmma_f32_16x16x32_f16(false, a, false, b, (short)0, c, false, false);
  }
  const float bias = b2[col];
  const bool full = (node0 + 16 <= n);
  float* obase = x + (size_t)(node0 + 8 * laneHi) * HDIM + col;
  if (full) {
#pragma unroll
    for (int r = 0; r < 8; r++)
      obase[(size_t)r * HDIM] = xs[r + 8 * laneHi][col] + c[r] + bias;
  } else {
#pragma unroll
    for (int r = 0; r < 8; r++)
      if (node0 + 8 * laneHi + r < n)
        obase[(size_t)r * HDIM] = xs[r + 8 * laneHi][col] + c[r] + bias;
  }
}

// -------------------- global max pool over batch segments --------------------

__global__ void pool_max_kernel(const float* __restrict__ x,
                                const int* __restrict__ batch,
                                float* __restrict__ pooled, int n) {
  int i = blockIdx.x * blockDim.x + threadIdx.x;
  if (i >= n * HDIM) return;
  int node = i >> 6, f = i & 63;
  atomic_max_f32(&pooled[batch[node] * HDIM + f], x[i]);
}

// -------------------- output MLP head + L2 normalize --------------------

__global__ __launch_bounds__(128)
void mlp_head_kernel(const float* __restrict__ pooled,
                     const float* __restrict__ Wo1, const float* __restrict__ bo1,
                     const float* __restrict__ Wo2, const float* __restrict__ bo2,
                     float* __restrict__ out) {
  __shared__ float pr[HDIM];
  __shared__ float t1[EMBD];
  __shared__ float red[EMBD];
  const int b = blockIdx.x;
  const int tid = threadIdx.x;
  if (tid < HDIM) pr[tid] = pooled[b * HDIM + tid];
  __syncthreads();

  float acc = bo1[tid];
  for (int j = 0; j < HDIM; j++) acc += pr[j] * Wo1[j * EMBD + tid];
  t1[tid] = fmaxf(acc, 0.f);
  __syncthreads();

  float acc2 = bo2[tid];
  for (int j = 0; j < EMBD; j++) acc2 += t1[j] * Wo2[j * EMBD + tid];
  red[tid] = acc2 * acc2;
  __syncthreads();
  for (int sft = 64; sft > 0; sft >>= 1) {
    if (tid < sft) red[tid] += red[tid + sft];
    __syncthreads();
  }
  float norm = sqrtf(red[0]);
  out[b * EMBD + tid] = acc2 / fmaxf(norm, 1e-12f);
}

// ---------------------------------------------------------------------------
// host launch
// ---------------------------------------------------------------------------

extern "C" void kernel_launch(void* const* d_in, const int* in_sizes, int n_in,
                              void* d_out, int out_size, void* d_ws, size_t ws_size,
                              hipStream_t stream) {
  const float* x_in  = (const float*)d_in[0];
  const float* ea    = (const float*)d_in[1];
  const int*   ei    = (const int*)d_in[2];
  const int*   batch = (const int*)d_in[3];
  const float* W_in  = (const float*)d_in[4];
  const float* b_in  = (const float*)d_in[5];
  const float* ln1_g = (const float*)d_in[6];
  const float* ln1_b = (const float*)d_in[7];
  const float* Wq = (const float*)d_in[8];   const float* bq = (const float*)d_in[9];
  const float* Wk = (const float*)d_in[10];  const float* bk = (const float*)d_in[11];
  const float* Wv = (const float*)d_in[12];  const float* bv = (const float*)d_in[13];
  const float* We = (const float*)d_in[14];
  const float* Wskip = (const float*)d_in[15]; const float* bskip = (const float*)d_in[16];
  const float* Wbeta = (const float*)d_in[17];
  const float* ln2_g = (const float*)d_in[18]; const float* ln2_b = (const float*)d_in[19];
  const float* W1 = (const float*)d_in[20];  const float* b1 = (const float*)d_in[21];
  const float* W2 = (const float*)d_in[22];  const float* b2 = (const float*)d_in[23];
  const float* Wo1 = (const float*)d_in[24]; const float* bo1 = (const float*)d_in[25];
  const float* Wo2 = (const float*)d_in[26]; const float* bo2 = (const float*)d_in[27];

  const int nN = in_sizes[0] / 8;       // 50000
  const int nE = in_sizes[1] / 4;       // 1000000
  const int nB = 512;
  const int L  = 3;

  // workspace layout (floats)
  float* ws = (float*)d_ws;
  float* xb   = ws;  ws += (size_t)nN * HDIM;
  float* qb   = ws;  ws += (size_t)nN * HDIM;
  float* kb   = ws;  ws += (size_t)nN * HDIM;
  float* vb   = ws;  ws += (size_t)nN * HDIM;
  float* xrb  = ws;  ws += (size_t)nN * HDIM;
  float* numb = ws;  ws += (size_t)nN * HDIM;
  float* mb   = ws;  ws += (size_t)nN * NHEAD;
  float* den  = ws;  ws += (size_t)nN * NHEAD;
  float* sb   = ws;  ws += (size_t)nE * NHEAD;
  float* pool = ws;  ws += (size_t)nB * HDIM;

  // f16 packed-weight region (32B aligned)
  uintptr_t pb = ((uintptr_t)ws + 31) & ~(uintptr_t)31;
  _Float16* packbase = (_Float16*)pb;
  const size_t PQKV = 4096;            // one 64x64 matrix packed
  const size_t PFFN = 8192;            // 64x128 or 128x64 packed
  const size_t PLYR = 4 * PQKV + 2 * PFFN;  // 32768 halfs / layer

  const int TB = 256;
  const int gNH  = (nN * HDIM + TB - 1) / TB;
  const int gNh4 = (nN * NHEAD + TB - 1) / TB;
  const int gE4  = (nE * NHEAD + TB - 1) / TB;
  const int gN   = (nN + TB - 1) / TB;
  const int gT16 = (nN + 15) / 16;
  const float NEG_INF = -__builtin_inff();

  // ---- one-shot weight repack (runs every call; trivial cost) ----
  for (int l = 0; l < L; l++) {
    _Float16* base = packbase + (size_t)l * PLYR;
    pack_b_kernel<<<1, 256, 0, stream>>>(Wq + (size_t)l * 4096, base + 0 * PQKV, 64, 64);
    pack_b_kernel<<<1, 256, 0, stream>>>(Wk + (size_t)l * 4096, base + 1 * PQKV, 64, 64);
    pack_b_kernel<<<1, 256, 0, stream>>>(Wv + (size_t)l * 4096, base + 2 * PQKV, 64, 64);
    pack_b_kernel<<<1, 256, 0, stream>>>(Wskip + (size_t)l * 4096, base + 3 * PQKV, 64, 64);
    pack_b_kernel<<<2, 256, 0, stream>>>(W1 + (size_t)l * 8192, base + 4 * PQKV, 64, 128);
    pack_b_kernel<<<2, 256, 0, stream>>>(W2 + (size_t)l * 8192, base + 4 * PQKV + PFFN, 128, 64);
  }

  input_proj_kernel<<<gNH, TB, 0, stream>>>(x_in, W_in, b_in, xb, nN);

  for (int l = 0; l < L; l++) {
    _Float16* base = packbase + (size_t)l * PLYR;

    ln_qkv_skip_kernel<<<gT16, 128, 0, stream>>>(
        xb, ln1_g + l * HDIM, ln1_b + l * HDIM,
        base + 0 * PQKV, bq + l * HDIM,
        base + 1 * PQKV, bk + l * HDIM,
        base + 2 * PQKV, bv + l * HDIM,
        base + 3 * PQKV, bskip + l * HDIM,
        qb, kb, vb, xrb, nN);

    fill_kernel<<<gNh4, TB, 0, stream>>>(mb, NEG_INF, nN * NHEAD);
    fill_kernel<<<gNh4, TB, 0, stream>>>(den, 0.f, nN * NHEAD);
    fill_kernel<<<gNH, TB, 0, stream>>>(numb, 0.f, nN * HDIM);

    edge_score_kernel<<<gE4, TB, 0, stream>>>(
        ei, ea, We + (size_t)l * 4 * HDIM, qb, kb, sb, mb, nE);
    fix_neginf_kernel<<<gNh4, TB, 0, stream>>>(mb, nN * NHEAD);
    edge_accum_kernel<<<gE4, TB, 0, stream>>>(
        ei, ea, We + (size_t)l * 4 * HDIM, vb, sb, mb, den, numb, nE);

    node_gate_kernel<<<gN, TB, 0, stream>>>(
        numb, den, xrb, Wbeta + (size_t)l * 3 * HDIM, xb, nN);

    ln_ffn_kernel<<<gT16, 128, 0, stream>>>(
        xb, ln2_g + l * HDIM, ln2_b + l * HDIM,
        base + 4 * PQKV, b1 + (size_t)l * 2 * HDIM,
        base + 4 * PQKV + PFFN, b2 + l * HDIM, nN);
  }

  fill_kernel<<<(nB * HDIM + TB - 1) / TB, TB, 0, stream>>>(pool, NEG_INF, nB * HDIM);
  pool_max_kernel<<<gNH, TB, 0, stream>>>(xb, batch, pool, nN);
  fix_neginf_kernel<<<(nB * HDIM + TB - 1) / TB, TB, 0, stream>>>(pool, nB * HDIM);

  mlp_head_kernel<<<nB, 128, 0, stream>>>(pool, Wo1, bo1, Wo2, bo2, (float*)d_out);
}
